// FILM_Encoder_SubGraph_59425167507617
// MI455X (gfx1250) — matrix operand order repaired and checked by reading, not compiled, not run on hardware
//
#include <hip/hip_runtime.h>
#include <hip/hip_bf16.h>
#include <math.h>

// ---------------- problem constants (from reference) ----------------
#define N_NODES 10000
#define E_EDGES 320000
#define IN_DIM  500
#define HID     320
#define NCLS    40
#define TOPK    5
#define BN_EPS  1e-5f

typedef __attribute__((ext_vector_type(16))) __bf16 v16bf;
typedef __attribute__((ext_vector_type(8)))  float  v8f;

union Frag16 { uint4 q[2]; v16bf v; };

static __device__ __forceinline__ unsigned short f2bf(float f) {
    unsigned int u = __float_as_uint(f);
    unsigned int r = (u + 0x7FFFu + ((u >> 16) & 1u)) >> 16;  // RNE
    return (unsigned short)r;
}

// ---------------- fp32 -> bf16 with K-padding ----------------
__global__ void cvt_pad_kernel(const float* __restrict__ in,
                               unsigned short* __restrict__ out,
                               int M, int Kin, int Kpad) {
    long tid = blockIdx.x * (long)blockDim.x + threadIdx.x;
    long total = (long)M * Kpad;
    if (tid >= total) return;
    int k = (int)(tid % Kpad);
    int m = (int)(tid / Kpad);
    out[tid] = (k < Kin) ? f2bf(in[(long)m * Kin + k]) : (unsigned short)0;
}

// ---------------- weight pack: W (din x cols, f32, row-major) -> WT (cols x dinPad, bf16) ----------------
__global__ void pack_wt_kernel(const float* __restrict__ W,
                               unsigned short* __restrict__ WT,
                               int din, int dinPad, int cols, int col_off) {
    long tid = blockIdx.x * (long)blockDim.x + threadIdx.x;
    long total = (long)cols * dinPad;
    if (tid >= total) return;
    int k = (int)(tid % dinPad);
    int j = (int)(tid / dinPad);
    unsigned short v = 0;
    if (k < din) v = f2bf(W[(long)k * cols + j]);
    WT[(long)(col_off + j) * dinPad + k] = v;
}

// ---------------- bf16 WMMA GEMM: C[MxN] = A[MxKp] * BT[NxKp]^T ----------------
// one wave computes a 16x64 tile (1 A-fragment reused over 4 B-fragments)
__global__ void __launch_bounds__(32)
wmma_gemm_kernel(const unsigned short* __restrict__ A,
                 const unsigned short* __restrict__ BT,
                 float* __restrict__ C,
                 int M, int Ncols, int Kpad) {
    int lane = threadIdx.x & 31;
    int hf   = lane >> 4;          // lane half selects K sub-range
    int lmod = lane & 15;
    int m0 = blockIdx.y * 16;
    int n0 = blockIdx.x * 64;

    v8f acc[4] = {};
    const unsigned short* arow = A + (long)(m0 + lmod) * Kpad;

    for (int ks = 0; ks < Kpad; ks += 32) {
        if (ks + 128 < Kpad)                      // WGP-scope prefetch of next A chunk
            __builtin_prefetch((const void*)(arow + ks + 128), 0, 3);
        Frag16 a;
        // A swizzle: lane<16 -> K {0..7,16..23}; lane>=16 -> K {8..15,24..31}
        a.q[0] = *(const uint4*)(arow + ks + hf * 8);
        a.q[1] = *(const uint4*)(arow + ks + 16 + hf * 8);
#pragma unroll
        for (int j = 0; j < 4; ++j) {
            int n = n0 + j * 16 + lmod;
            if (n >= Ncols) n = Ncols - 1;              // clamp (store is guarded)
            const unsigned short* brow = BT + (long)n * Kpad + ks + hf * 8;
            Frag16 b;                                    // B: same K swizzle as A
            b.q[0] = *(const uint4*)(brow);
            b.q[1] = *(const uint4*)(brow + 16);
            acc[j] = __builtin_amdgcn_wmma_f32_16x16x32_bf16(
                false, a.v, false, b.v, (short)0, acc[j], false, false);
        }
    }
#pragma unroll
    for (int j = 0; j < 4; ++j) {
        int n = n0 + j * 16 + lmod;
        if (n >= Ncols) continue;
#pragma unroll
        for (int r = 0; r < 8; ++r) {
            int m = m0 + r + hf * 8;                     // C/D layout
            C[(long)m * Ncols + n] = acc[j][r];
        }
    }
}

// ---------------- FiLM edge messages with atomic segment-sum ----------------
__global__ void edge_msg_kernel(const float* __restrict__ Z, const int* __restrict__ ei,
                                float* __restrict__ agg, int d, int ldz, int act) {
    long tid = blockIdx.x * (long)blockDim.x + threadIdx.x;
    int per_edge = d >> 2;
    long total = (long)E_EDGES * per_edge;
    if (tid >= total) return;
    int e  = (int)(tid / per_edge);
    int j0 = (int)(tid % per_edge) * 4;
    int src = ei[e];
    int dst = ei[E_EDGES + e];
    const float* zs = Z + (long)src * ldz;   // lin at offset 0
    const float* zd = Z + (long)dst * ldz;   // beta at d, gamma at 2d
#pragma unroll
    for (int t = 0; t < 4; ++t) {
        int j = j0 + t;
        float m = zd[2 * d + j] * zs[j] + zd[d + j];
        if (act) m = fmaxf(m, 0.f);
        atomicAdd(&agg[(long)dst * d + j], m);
    }
}

__global__ void edge_cnt_kernel(const int* __restrict__ ei, float* __restrict__ cnt) {
    int e = blockIdx.x * blockDim.x + threadIdx.x;
    if (e < E_EDGES) atomicAdd(&cnt[ei[E_EDGES + e]], 1.f);
}

// ---------------- node self term + mean-agg + optional BN ----------------
__global__ void combine_kernel(const float* __restrict__ Z, const float* __restrict__ agg,
                               const float* __restrict__ cnt, float* __restrict__ out,
                               int d, int ldz, int act, int use_bn,
                               const float* __restrict__ bw, const float* __restrict__ bb,
                               const float* __restrict__ brm, const float* __restrict__ brv) {
    long tid = blockIdx.x * (long)blockDim.x + threadIdx.x;
    long total = (long)N_NODES * d;
    if (tid >= total) return;
    int j = (int)(tid % d);
    int n = (int)(tid / d);
    const float* z = Z + (long)n * ldz;
    float o = z[5 * d + j] * z[3 * d + j] + z[4 * d + j];  // fs_gamma*skip + fs_beta
    if (act) o = fmaxf(o, 0.f);
    float h = o + agg[tid] / fmaxf(cnt[n], 1.f);
    if (use_bn) h = (h - brm[j]) * rsqrtf(brv[j] + BN_EPS) * bw[j] + bb[j];
    out[tid] = h;
}

// ---------------- L2-normalize embedding rows -> bf16 ----------------
__global__ void __launch_bounds__(32)
norm_kernel(const float* __restrict__ emb, unsigned short* __restrict__ EN) {
    int row = blockIdx.x;
    int lane = threadIdx.x;
    const float* e = emb + (long)row * HID;
    float s = 0.f;
    for (int j = lane; j < HID; j += 32) { float v = e[j]; s += v * v; }
#pragma unroll
    for (int off = 16; off > 0; off >>= 1) s += __shfl_xor(s, off, 32);
    float inv = 1.f / fmaxf(sqrtf(s), 1e-8f);
    for (int j = lane; j < HID; j += 32) EN[(long)row * HID + j] = f2bf(e[j] * inv);
}

// ---------------- fused en @ en^T + per-row top-5 ----------------
// 1 wave per 16 rows. A fragments pinned in VGPRs; B tiles (16 x HID bf16 = 10KB)
// double-buffered in LDS via global_load_async_to_lds_b128 (ASYNCcnt pipeline).
#define B_TILE_BYTES  (16 * HID * 2)          // 10240 B
#define B_TILE_CHUNKS (B_TILE_BYTES / 16)     // 640 x b128
#define CHUNKS_PER_ROW (HID * 2 / 16)         // 40

__global__ void __launch_bounds__(32)
sim_topk_kernel(const unsigned short* __restrict__ EN,
                float* __restrict__ topv, int* __restrict__ topi) {
    __shared__ __align__(16) unsigned short bufB[2][16 * HID];
    __shared__ float tile[16 * 16];
    int lane = threadIdx.x & 31;
    int hf   = lane >> 4;
    int lmod = lane & 15;
    int r0 = blockIdx.x * 16;

    // keep all 10 K-fragments of the 16-row A tile resident in VGPRs
    Frag16 aF[HID / 32];
    const unsigned short* arow = EN + (long)(r0 + lmod) * HID;
#pragma unroll
    for (int kt = 0; kt < HID / 32; ++kt) {
        aF[kt].q[0] = *(const uint4*)(arow + kt * 32 + hf * 8);
        aF[kt].q[1] = *(const uint4*)(arow + kt * 32 + 16 + hf * 8);
    }

    // async-stage one 16-row B tile into LDS buffer `buf` (20 b128 chunks per lane)
    auto issue_tile = [&](int c0, int buf) {
        unsigned ldsBase = (unsigned)(uintptr_t)&bufB[buf][0];
#pragma unroll
        for (int i = 0; i < B_TILE_CHUNKS / 32; ++i) {
            int ci  = i * 32 + lane;                 // chunk id 0..639
            int row = ci / CHUNKS_PER_ROW;
            int off = (ci % CHUNKS_PER_ROW) * 8;     // ushort offset within row
            unsigned long long src =
                (unsigned long long)(uintptr_t)(EN + (long)(c0 + row) * HID + off);
            unsigned dst = ldsBase + (unsigned)ci * 16;
            asm volatile("global_load_async_to_lds_b128 %0, %1, off"
                         :: "v"(dst), "v"(src) : "memory");
        }
    };

    float t0 = -1e30f, t1 = -1e30f, t2 = -1e30f, t3 = -1e30f, t4 = -1e30f;
    int   i0 = 0, i1 = 0, i2 = 0, i3 = 0, i4 = 0;

    issue_tile(0, 0);
    for (int c0 = 0; c0 < N_NODES; c0 += 16) {
        int buf = (c0 >> 4) & 1;
        if (c0 + 16 < N_NODES) {
            issue_tile(c0 + 16, buf ^ 1);
            // async loads complete in order: <=20 outstanding ==> current tile resident
            asm volatile("s_wait_asynccnt 20" ::: "memory");
        } else {
            asm volatile("s_wait_asynccnt 0" ::: "memory");
        }

        v8f acc = {};
        const unsigned short* brow = &bufB[buf][lmod * HID];
#pragma unroll
        for (int kt = 0; kt < HID / 32; ++kt) {
            Frag16 b;                                 // ds_load_b128 x2 from staged tile
            b.q[0] = *(const uint4*)(brow + kt * 32 + hf * 8);
            b.q[1] = *(const uint4*)(brow + kt * 32 + 16 + hf * 8);
            acc = __builtin_amdgcn_wmma_f32_16x16x32_bf16(
                false, aF[kt].v, false, b.v, (short)0, acc, false, false);
        }
        __syncthreads();
#pragma unroll
        for (int r = 0; r < 8; ++r) tile[(r + hf * 8) * 16 + lmod] = acc[r];
        __syncthreads();
        if (lane < 16) {
            for (int c = 0; c < 16; ++c) {
                float v = tile[lane * 16 + c];
                if (v > t4) {
                    int idx = c0 + c;
                    t4 = v; i4 = idx;
                    if (t4 > t3) { float tv = t3; t3 = t4; t4 = tv; int ti = i3; i3 = i4; i4 = ti; }
                    if (t3 > t2) { float tv = t2; t2 = t3; t3 = tv; int ti = i2; i2 = i3; i3 = ti; }
                    if (t2 > t1) { float tv = t1; t1 = t2; t2 = tv; int ti = i1; i1 = i2; i2 = ti; }
                    if (t1 > t0) { float tv = t0; t0 = t1; t1 = tv; int ti = i0; i0 = i1; i1 = ti; }
                }
            }
        }
        __syncthreads();
    }
    if (lane < 16) {
        int row = r0 + lane;
        topv[row * TOPK + 0] = t0; topi[row * TOPK + 0] = i0;
        topv[row * TOPK + 1] = t1; topi[row * TOPK + 1] = i1;
        topv[row * TOPK + 2] = t2; topi[row * TOPK + 2] = i2;
        topv[row * TOPK + 3] = t3; topi[row * TOPK + 3] = i3;
        topv[row * TOPK + 4] = t4; topi[row * TOPK + 4] = i4;
    }
}

// ---------------- final: eta*log_softmax(lc) + (1-eta)*log_softmax(fused) ----------------
__global__ void final_kernel(const float* __restrict__ lc, const float* __restrict__ topv,
                             const int* __restrict__ topi, const int* __restrict__ y,
                             const float* __restrict__ eta_p, float* __restrict__ out) {
    int n = blockIdx.x * blockDim.x + threadIdx.x;
    if (n >= N_NODES) return;
    float eta = eta_p[0];
    const float* l = lc + (long)n * NCLS;

    float mx = -1e30f;
#pragma unroll
    for (int c = 0; c < NCLS; ++c) mx = fmaxf(mx, l[c]);
    float s = 0.f;
#pragma unroll
    for (int c = 0; c < NCLS; ++c) s += expf(l[c] - mx);
    float lse = mx + logf(s);

    float ev[TOPK]; int yy[TOPK];
#pragma unroll
    for (int k = 0; k < TOPK; ++k) {
        ev[k] = expf(topv[n * TOPK + k]);
        yy[k] = y[topi[n * TOPK + k]];
    }
    float fr[NCLS];
    float mx2 = -1e30f;
#pragma unroll
    for (int c = 0; c < NCLS; ++c) {
        float f = 0.f;
#pragma unroll
        for (int k = 0; k < TOPK; ++k) f += (yy[k] == c) ? ev[k] : 0.f;
        fr[c] = f;
        mx2 = fmaxf(mx2, f);
    }
    float s2 = 0.f;
#pragma unroll
    for (int c = 0; c < NCLS; ++c) s2 += expf(fr[c] - mx2);
    float lse2 = mx2 + logf(s2);

    float* o = out + (long)n * NCLS;
#pragma unroll
    for (int c = 0; c < NCLS; ++c)
        o[c] = eta * (l[c] - lse) + (1.f - eta) * (fr[c] - lse2);
}

// =====================================================================
extern "C" void kernel_launch(void* const* d_in, const int* in_sizes, int n_in,
                              void* d_out, int out_size, void* d_ws, size_t ws_size,
                              hipStream_t stream) {
    const float* x  = (const float*)d_in[0];
    const int*   ei = (const int*)d_in[1];
    const int*   y  = (const int*)d_in[2];
    // d_in[3] = mask (unused by reference)
    const float* W[4][4];
    for (int l = 0; l < 4; ++l)
        for (int j = 0; j < 4; ++j)
            W[l][j] = (const float*)d_in[4 + 4 * l + j];
    const float* BN[3][4];
    for (int l = 0; l < 3; ++l)
        for (int j = 0; j < 4; ++j)
            BN[l][j] = (const float*)d_in[20 + 4 * l + j];
    const float* eta = (const float*)d_in[32];

    float* out_final = (float*)d_out;
    float* emb       = (float*)d_out + (size_t)N_NODES * NCLS;   // (final, embedding)

    const int din[4]    = {IN_DIM, HID, HID, HID};
    const int dinPad[4] = {512, HID, HID, HID};
    const int dout[4]   = {HID, HID, HID, NCLS};

    // ---- workspace bump allocator ----
    size_t off = 0;
    auto bump = [&](size_t bytes) -> void* {
        void* p = (char*)d_ws + off;
        off += (bytes + 255) & ~(size_t)255;
        return p;
    };
    unsigned short* WT[4];
    for (int l = 0; l < 4; ++l)
        WT[l] = (unsigned short*)bump((size_t)6 * dout[l] * dinPad[l] * sizeof(unsigned short));
    unsigned short* hA = (unsigned short*)bump((size_t)N_NODES * 512 * sizeof(unsigned short));
    float* Z   = (float*)bump((size_t)N_NODES * 6 * HID * sizeof(float));
    float* agg = (float*)bump((size_t)N_NODES * HID * sizeof(float));
    float* cnt = (float*)bump((size_t)N_NODES * sizeof(float));
    float* h1  = (float*)bump((size_t)N_NODES * HID * sizeof(float));
    float* h2  = (float*)bump((size_t)N_NODES * HID * sizeof(float));
    float* lc  = (float*)bump((size_t)N_NODES * NCLS * sizeof(float));
    unsigned short* EN = (unsigned short*)bump((size_t)N_NODES * HID * sizeof(unsigned short));
    float* topv = (float*)bump((size_t)N_NODES * TOPK * sizeof(float));
    int*   topi = (int*)bump((size_t)N_NODES * TOPK * sizeof(int));

    // ---- pack weights into transposed bf16 (cols x dinPad): [lin|film(2d)|skip|film_skip(2d)] ----
    for (int l = 0; l < 4; ++l) {
        int d = dout[l];
        const int cols[4] = {d, 2 * d, d, 2 * d};
        const int coff[4] = {0, d, 3 * d, 4 * d};
        for (int p = 0; p < 4; ++p) {
            long total = (long)cols[p] * dinPad[l];
            int blocks = (int)((total + 255) / 256);
            pack_wt_kernel<<<blocks, 256, 0, stream>>>(W[l][p], WT[l], din[l], dinPad[l],
                                                       cols[p], coff[p]);
        }
    }

    // ---- per-layer FiLM conv ----
    const float* in_ptr[4]  = {x, h1, h2, emb};
    float*       out_ptr[4] = {h1, h2, emb, lc};
    for (int l = 0; l < 4; ++l) {
        int d = dout[l], Kp = dinPad[l], Ncol = 6 * d;
        int act = (l < 3), use_bn = (l < 3);

        long tc = (long)N_NODES * Kp;
        cvt_pad_kernel<<<(int)((tc + 255) / 256), 256, 0, stream>>>(in_ptr[l], hA, N_NODES,
                                                                    din[l], Kp);
        dim3 gg((Ncol + 63) / 64, N_NODES / 16);
        wmma_gemm_kernel<<<gg, 32, 0, stream>>>(hA, WT[l], Z, N_NODES, Ncol, Kp);

        hipMemsetAsync(agg, 0, (size_t)N_NODES * d * sizeof(float), stream);
        hipMemsetAsync(cnt, 0, (size_t)N_NODES * sizeof(float), stream);

        long te = (long)E_EDGES * (d >> 2);
        edge_msg_kernel<<<(int)((te + 255) / 256), 256, 0, stream>>>(Z, ei, agg, d, Ncol, act);
        edge_cnt_kernel<<<(E_EDGES + 255) / 256, 256, 0, stream>>>(ei, cnt);

        long tn = (long)N_NODES * d;
        combine_kernel<<<(int)((tn + 255) / 256), 256, 0, stream>>>(
            Z, agg, cnt, out_ptr[l], d, Ncol, act, use_bn,
            use_bn ? BN[l][0] : nullptr, use_bn ? BN[l][1] : nullptr,
            use_bn ? BN[l][2] : nullptr, use_bn ? BN[l][3] : nullptr);
    }

    // ---- similarity head ----
    norm_kernel<<<N_NODES, 32, 0, stream>>>(emb, EN);
    sim_topk_kernel<<<N_NODES / 16, 32, 0, stream>>>(EN, topv, topi);
    final_kernel<<<(N_NODES + 255) / 256, 256, 0, stream>>>(lc, topv, topi, y, eta, out_final);
}
